// CausalAttention_40192303956199
// MI455X (gfx1250) — matrix operand order repaired and checked by reading, not compiled, not run on hardware
//
#include <hip/hip_runtime.h>
#include <hip/hip_bf16.h>

typedef __attribute__((ext_vector_type(4)))  __bf16 v4bf;
typedef __attribute__((ext_vector_type(8)))  __bf16 v8bf;
typedef __attribute__((ext_vector_type(16))) __bf16 v16bf;
typedef __attribute__((ext_vector_type(8)))  float  v8f;

#define B_   4
#define T_   2048
#define D_   1024
#define H_   16
#define HD_  64
#define QKVN (3 * D_)

// Padded LDS row strides (elements) to spread banks; rows stay 16B aligned.
#define APAD 40   // 80B rows
#define KPAD 72   // 144B rows

__device__ __forceinline__ v16bf combine16(v8bf lo, v8bf hi) {
  return __builtin_shufflevector(lo, hi, 0, 1, 2, 3, 4, 5, 6, 7,
                                 8, 9, 10, 11, 12, 13, 14, 15);
}

// Async global->LDS 8-byte copy (CDNA5 GLOBAL_LOAD_ASYNC_TO_LDS_B64).
// ldsaddr: LDS byte address (low 32 bits of generic pointer), gaddr: 64-bit
// global address. Tracked by ASYNCcnt; pair with async_wait() before barrier.
__device__ __forceinline__ void async_copy_b64(unsigned ldsaddr,
                                               const void* gaddr) {
  asm volatile("global_load_async_to_lds_b64 %0, %1, off"
               :
               : "v"(ldsaddr), "v"((unsigned long long)(uintptr_t)gaddr)
               : "memory");
}
__device__ __forceinline__ void async_wait() {
  asm volatile("s_wait_asynccnt 0" ::: "memory");
}
__device__ __forceinline__ unsigned lds_addr32(const void* p) {
  return (unsigned)(uintptr_t)p;  // low 32 bits serve as the LDS address
}

__global__ void cvt_f32_bf16(const float* __restrict__ in,
                             __bf16* __restrict__ out, int n) {
  int i = blockIdx.x * blockDim.x + threadIdx.x;
  int stride = gridDim.x * blockDim.x;
  for (; i < n; i += stride) out[i] = (__bf16)in[i];
}

// Stage one 128x32 A tile (async copies) and one 32x128 B tile (transposed,
// via VGPRs) into padded LDS buffers. 128 threads.
__device__ __forceinline__ void stage_tiles(const __bf16* __restrict__ A,
                                            const __bf16* __restrict__ Bm,
                                            __bf16* bufA, __bf16* bufBT,
                                            int m0, int n0, int k0, int K,
                                            int N, int tid) {
#pragma unroll
  for (int it = 0; it < 8; ++it) {
    int idx = it * 128 + tid;  // 0..1023 u64 (128 rows x 8 u64)
    int m = idx >> 3;
    int q = idx & 7;
    async_copy_b64(lds_addr32(((uint64_t*)bufA) + m * (APAD / 4) + q),
                   ((const uint64_t*)(A + (size_t)(m0 + m) * K + k0)) + q);
  }
#pragma unroll
  for (int it = 0; it < 8; ++it) {
    int idx = it * 128 + tid;  // 0..1023 quads (32 k x 32 n-quads)
    int n4 = idx & 31;
    int k = idx >> 5;
    v4bf v = ((const v4bf*)(Bm + (size_t)(k0 + k) * N + n0))[n4];
#pragma unroll
    for (int j = 0; j < 4; ++j) bufBT[(n4 * 4 + j) * APAD + k] = v[j];
  }
}

// C = A(bf16,[M,K]) x B(bf16,[K,N]); fp32 accum.
// Block: 128 threads (4 waves), tile 128x128, BK=32, double-buffered LDS.
// Wave grid 2x2: wave tile 64x64 = 4 m-tiles x 4 n-tiles (16 independent accums).
template <bool OUT_BF16>
__global__ __launch_bounds__(128) void gemm_bf16(
    const __bf16* __restrict__ A, const __bf16* __restrict__ Bm,
    void* __restrict__ Out, int M, int N, int K) {
  __shared__ __align__(128) __bf16 ldsA[2][128 * APAD];
  __shared__ __align__(128) __bf16 ldsBT[2][128 * APAD];

  const int tid  = threadIdx.x;
  const int lane = tid & 31;
  const int wid  = tid >> 5;
  const int wm   = wid >> 1;   // 0..1
  const int wn   = wid & 1;    // 0..1
  const int hl   = lane >> 4;
  const int ln   = lane & 15;

  const int m0 = blockIdx.y * 128;
  const int n0 = blockIdx.x * 128;

  const v8f vzero = {0.f, 0.f, 0.f, 0.f, 0.f, 0.f, 0.f, 0.f};
  v8f acc[4][4];
#pragma unroll
  for (int mt = 0; mt < 4; ++mt)
#pragma unroll
    for (int nt = 0; nt < 4; ++nt) acc[mt][nt] = vzero;

  const int nk = K >> 5;
  stage_tiles(A, Bm, ldsA[0], ldsBT[0], m0, n0, 0, K, N, tid);
  async_wait();
  __syncthreads();

  for (int kt = 0; kt < nk; ++kt) {
    const __bf16* bA = ldsA[kt & 1];
    const __bf16* bB = ldsBT[kt & 1];

    v16bf aF[4], bF[4];
#pragma unroll
    for (int mt = 0; mt < 4; ++mt) {
      const __bf16* rp = bA + (size_t)(wm * 64 + mt * 16 + ln) * APAD;
      v8bf lo = *(const v8bf*)(rp + hl * 16);
      v8bf hi = *(const v8bf*)(rp + hl * 16 + 8);
      aF[mt] = combine16(lo, hi);
    }
#pragma unroll
    for (int nt = 0; nt < 4; ++nt) {
      const __bf16* cp = bB + (size_t)(wn * 64 + nt * 16 + ln) * APAD + hl * 16;
      v8bf lo = *(const v8bf*)(cp);
      v8bf hi = *(const v8bf*)(cp + 8);
      bF[nt] = combine16(lo, hi);
    }

    if (kt + 1 < nk)
      stage_tiles(A, Bm, ldsA[(kt + 1) & 1], ldsBT[(kt + 1) & 1], m0, n0,
                  (kt + 1) * 32, K, N, tid);

#pragma unroll
    for (int mt = 0; mt < 4; ++mt)
#pragma unroll
      for (int nt = 0; nt < 4; ++nt)
        acc[mt][nt] = __builtin_amdgcn_wmma_f32_16x16x32_bf16(
            false, aF[mt], false, bF[nt], (short)0, acc[mt][nt], false, false);

    async_wait();  // this wave's async tile copies landed in LDS
    __syncthreads();
  }

#pragma unroll
  for (int mt = 0; mt < 4; ++mt) {
    int row = m0 + wm * 64 + mt * 16 + hl * 8;
#pragma unroll
    for (int nt = 0; nt < 4; ++nt) {
      int col = n0 + wn * 64 + nt * 16 + ln;
#pragma unroll
      for (int r = 0; r < 8; ++r) {
        float v = acc[mt][nt][r];
        if (OUT_BF16)
          ((__bf16*)Out)[(size_t)(row + r) * N + col] = (__bf16)v;
        else
          ((float*)Out)[(size_t)(row + r) * N + col] = v;
      }
    }
  }
}

// Flash-style causal attention. qkv: bf16 [B*T, 3072] rows = Q|K|V.
// Block = 256 threads (8 waves); each wave: 16 query rows; block: 128 rows.
__global__ __launch_bounds__(256) void attn_kernel(
    const __bf16* __restrict__ qkv, __bf16* __restrict__ outA) {
  __shared__ __align__(128) __bf16 ldsK[32 * KPAD];    // [key][d] padded
  __shared__ __align__(128) __bf16 ldsVT[64 * APAD];   // [d][key] padded
  __shared__ __align__(128) __bf16 pbuf[8][16 * APAD]; // per-wave P tiles

  const int tid  = threadIdx.x;
  const int lane = tid & 31;
  const int wave = tid >> 5;   // 0..7
  const int hl   = lane >> 4;
  const int ln   = lane & 15;

  const int q0 = blockIdx.x * 128;
  const int bh = blockIdx.y;
  const int b  = bh >> 4;
  const int h  = bh & 15;

  const size_t rs = (size_t)QKVN;
  const __bf16* qkv_b = qkv + (size_t)b * T_ * rs;

  // Q A-fragments (HD=64 -> two K=32 fragments), loaded once.
  v16bf qA[2];
  {
    int trow = q0 + wave * 16 + ln;
    const __bf16* qp = qkv_b + (size_t)trow * rs + h * HD_;
#pragma unroll
    for (int f = 0; f < 2; ++f) {
      v8bf lo = *(const v8bf*)(qp + f * 32 + hl * 16);
      v8bf hi = *(const v8bf*)(qp + f * 32 + hl * 16 + 8);
      qA[f] = combine16(lo, hi);
    }
  }

  const v8f vzero = {0.f, 0.f, 0.f, 0.f, 0.f, 0.f, 0.f, 0.f};
  v8f o[4];
#pragma unroll
  for (int nt = 0; nt < 4; ++nt) o[nt] = vzero;
  float mrow[8], lrow[8];
#pragma unroll
  for (int r = 0; r < 8; ++r) { mrow[r] = -INFINITY; lrow[r] = 0.f; }

  const int kend = q0 + 128;
  const int qwave_hi = q0 + wave * 16 + 15;

  for (int k0 = 0; k0 < kend; k0 += 32) {
    // Stage K tile 32x64 via async global->LDS copies (padded rows)
#pragma unroll
    for (int it = 0; it < 2; ++it) {
      int idx = it * 256 + tid;  // 0..511 (32 rows x 16 u64)
      int kr = idx >> 4;
      int q8 = idx & 15;
      async_copy_b64(
          lds_addr32(((uint64_t*)ldsK) + kr * (KPAD / 4) + q8),
          ((const uint64_t*)(qkv_b + (size_t)(k0 + kr) * rs + D_ + h * HD_)) +
              q8);
    }
    // Stage V tile 32x64 transposed -> ldsVT[d][key] (vectorized reads)
#pragma unroll
    for (int it = 0; it < 2; ++it) {
      int idx = it * 256 + tid;  // 0..511 quads (32 k x 16 d-quads)
      int d4 = idx & 15;
      int kr = idx >> 4;
      v4bf v =
          ((const v4bf*)(qkv_b + (size_t)(k0 + kr) * rs + 2 * D_ + h * HD_))[d4];
#pragma unroll
      for (int j = 0; j < 4; ++j) ldsVT[(d4 * 4 + j) * APAD + kr] = v[j];
    }
    async_wait();
    __syncthreads();

    if (k0 <= qwave_hi) {  // wave-uniform skip of fully-masked steps
      v8f c0 = vzero, c1 = vzero;
#pragma unroll
      for (int f = 0; f < 2; ++f) {  // interleave two independent chains
        const __bf16* kp0 = ldsK + (size_t)ln * KPAD + f * 32 + hl * 16;
        v16bf b0 = combine16(*(const v8bf*)kp0, *(const v8bf*)(kp0 + 8));
        const __bf16* kp1 = ldsK + (size_t)(16 + ln) * KPAD + f * 32 + hl * 16;
        v16bf b1 = combine16(*(const v8bf*)kp1, *(const v8bf*)(kp1 + 8));
        c0 = __builtin_amdgcn_wmma_f32_16x16x32_bf16(false, qA[f], false, b0,
                                                     (short)0, c0, false, false);
        c1 = __builtin_amdgcn_wmma_f32_16x16x32_bf16(false, qA[f], false, b1,
                                                     (short)0, c1, false, false);
      }

      const float scale = 0.125f;  // 1/sqrt(64)
      const int key0 = k0 + ln;
      const int key1 = k0 + 16 + ln;
#pragma unroll
      for (int r = 0; r < 8; ++r) {
        int row = q0 + wave * 16 + hl * 8 + r;
        float s0 = (key0 <= row) ? c0[r] * scale : -INFINITY;
        float s1 = (key1 <= row) ? c1[r] * scale : -INFINITY;
        float mx = fmaxf(s0, s1);
        mx = fmaxf(mx, __shfl_xor(mx, 1, 16));
        mx = fmaxf(mx, __shfl_xor(mx, 2, 16));
        mx = fmaxf(mx, __shfl_xor(mx, 4, 16));
        mx = fmaxf(mx, __shfl_xor(mx, 8, 16));
        float mn = fmaxf(mrow[r], mx);
        float alpha = __expf(mrow[r] - mn);
        float p0 = __expf(s0 - mn);
        float p1 = __expf(s1 - mn);
        float prs = p0 + p1;
        prs += __shfl_xor(prs, 1, 16);
        prs += __shfl_xor(prs, 2, 16);
        prs += __shfl_xor(prs, 4, 16);
        prs += __shfl_xor(prs, 8, 16);
        lrow[r] = lrow[r] * alpha + prs;
        mrow[r] = mn;
#pragma unroll
        for (int nt = 0; nt < 4; ++nt) o[nt][r] *= alpha;
        __bf16* pr = &pbuf[wave][(hl * 8 + r) * APAD];
        pr[ln]      = (__bf16)p0;
        pr[16 + ln] = (__bf16)p1;
      }
      asm volatile("s_wait_dscnt 0" ::: "memory");  // intra-wave LDS bounce

      v16bf pA;
      {
        const __bf16* pp = &pbuf[wave][(size_t)ln * APAD];
        v8bf lo = *(const v8bf*)(pp + hl * 16);
        v8bf hi = *(const v8bf*)(pp + hl * 16 + 8);
        pA = combine16(lo, hi);
      }
#pragma unroll
      for (int nt = 0; nt < 4; ++nt) {
        const __bf16* vp = ldsVT + (size_t)(nt * 16 + ln) * APAD + hl * 16;
        v16bf vB = combine16(*(const v8bf*)vp, *(const v8bf*)(vp + 8));
        o[nt] = __builtin_amdgcn_wmma_f32_16x16x32_bf16(false, pA, false, vB,
                                                        (short)0, o[nt], false,
                                                        false);
      }
    }
    __syncthreads();
  }

#pragma unroll
  for (int r = 0; r < 8; ++r) {
    int trow = q0 + wave * 16 + hl * 8 + r;
    float inv = 1.0f / lrow[r];
#pragma unroll
    for (int nt = 0; nt < 4; ++nt) {
      int col = h * HD_ + nt * 16 + ln;
      outA[(size_t)(b * T_ + trow) * D_ + col] = (__bf16)(o[nt][r] * inv);
    }
  }
}

extern "C" void kernel_launch(void* const* d_in, const int* in_sizes, int n_in,
                              void* d_out, int out_size, void* d_ws,
                              size_t ws_size, hipStream_t stream) {
  (void)in_sizes; (void)n_in; (void)out_size; (void)ws_size;
  const float* x    = (const float*)d_in[0];
  const float* Wqkv = (const float*)d_in[1];
  const float* Wout = (const float*)d_in[2];

  char* ws = (char*)d_ws;
  size_t off = 0;
  __bf16* x_bf    = (__bf16*)(ws + off); off += (size_t)B_ * T_ * D_ * 2;
  __bf16* wqkv_bf = (__bf16*)(ws + off); off += (size_t)D_ * QKVN * 2;
  __bf16* wout_bf = (__bf16*)(ws + off); off += (size_t)D_ * D_ * 2;
  __bf16* qkv_bf  = (__bf16*)(ws + off); off += (size_t)B_ * T_ * QKVN * 2;
  __bf16* attn_bf = (__bf16*)(ws + off);

  const int n1 = B_ * T_ * D_;
  const int n2 = D_ * QKVN;
  const int n3 = D_ * D_;
  cvt_f32_bf16<<<(n1 + 255) / 256, 256, 0, stream>>>(x, x_bf, n1);
  cvt_f32_bf16<<<(n2 + 255) / 256, 256, 0, stream>>>(Wqkv, wqkv_bf, n2);
  cvt_f32_bf16<<<(n3 + 255) / 256, 256, 0, stream>>>(Wout, wout_bf, n3);

  dim3 g1(QKVN / 128, (B_ * T_) / 128);
  gemm_bf16<true><<<g1, 128, 0, stream>>>(x_bf, wqkv_bf, (void*)qkv_bf,
                                          B_ * T_, QKVN, D_);

  dim3 ga(T_ / 128, B_ * H_);
  attn_kernel<<<ga, 256, 0, stream>>>(qkv_bf, attn_bf);

  dim3 g2(D_ / 128, (B_ * T_) / 128);
  gemm_bf16<false><<<g2, 128, 0, stream>>>(attn_bf, wout_bf, d_out, B_ * T_,
                                           D_, D_);
}
